// MultiHeadAttention_62904091017750
// MI455X (gfx1250) — matrix operand order repaired and checked by reading, not compiled, not run on hardware
//
#include <hip/hip_runtime.h>
#include <hip/hip_bf16.h>

// ---------------------------------------------------------------------------
// MultiHeadAttention (d=1024, H=16, dk=64, L=2048, tile-mask window 512)
// bf16 WMMA pipeline: convert -> QKV GEMMs -> flash attention -> out GEMM
// Staging uses GLOBAL_LOAD_ASYNC_TO_LDS_B128 (ASYNCcnt) when available.
// ---------------------------------------------------------------------------

typedef __bf16 bf16;
typedef __attribute__((ext_vector_type(16))) __bf16 v16bf;
typedef __attribute__((ext_vector_type(8)))  __bf16 v8bf;
typedef __attribute__((ext_vector_type(8)))  float  v8f;
typedef __attribute__((ext_vector_type(4)))  int    v4i;

#define AS1 __attribute__((address_space(1)))
#define AS3 __attribute__((address_space(3)))

#define L_SEQ 2048
#define DM    1024
#define NH    16
#define DKH   64
#define WIN   512

#if defined(__gfx1250__) && __has_builtin(__builtin_amdgcn_global_load_async_to_lds_b128)
#define HAVE_ASYNC_LDS 1
#else
#define HAVE_ASYNC_LDS 0
#endif

__device__ __forceinline__ void copy16_g2lds(const bf16* g, bf16* l) {
#if HAVE_ASYNC_LDS
  // Prototype (from hipcc diagnostic): (int4 __device__*, int4 __shared__*, imm, imm)
  __builtin_amdgcn_global_load_async_to_lds_b128(
      (AS1 v4i*)g, (AS3 v4i*)l, 0, 0);
#else
  *(uint4*)l = *(const uint4*)g;
#endif
}

__device__ __forceinline__ void wait_async_lds() {
#if HAVE_ASYNC_LDS
  asm volatile("s_wait_asynccnt 0x0" ::: "memory");
#endif
}

__device__ __forceinline__ bf16 to_bf16(float f) {
  union { float f; unsigned u; } a; a.f = f;
  unsigned r = a.u + 0x7FFFu + ((a.u >> 16) & 1u);  // round-to-nearest-even
  unsigned short h = (unsigned short)(r >> 16);
  return __builtin_bit_cast(bf16, h);
}

__device__ __forceinline__ v8f zero_v8f() {
  v8f z;
#pragma unroll
  for (int i = 0; i < 8; ++i) z[i] = 0.f;
  return z;
}

// A-operand (16x32 MxK, 16-bit): lanes 0-15 hold K 0-15, lanes 16-31
// hold K 16-31 (one contiguous 32B chunk per lane). `row` -> K=0 elem.
__device__ __forceinline__ v16bf load_a_frag(const bf16* row, int hi) {
  const v8bf lo = *(const v8bf*)(row + (hi ? 16 : 0));
  const v8bf hv = *(const v8bf*)(row + (hi ? 24 : 8));
  v16bf r;
#pragma unroll
  for (int e = 0; e < 8; ++e) { r[e] = lo[e]; r[e + 8] = hv[e]; }
  return r;
}

// B-operand (32x16 KxN, 16-bit): lane n holds column n; lanes 0-15 K=0-15,
// lanes 16-31 K=16-31 (one contiguous 32B chunk). `row` -> column n, K=0 elem.
__device__ __forceinline__ v16bf load_b_frag(const bf16* row, int hi) {
  const bf16* p = row + (hi ? 16 : 0);
  const v8bf lo = *(const v8bf*)(p);
  const v8bf hv = *(const v8bf*)(p + 8);
  v16bf r;
#pragma unroll
  for (int e = 0; e < 8; ++e) { r[e] = lo[e]; r[e + 8] = hv[e]; }
  return r;
}

__device__ __forceinline__ v8f wmma_bf(v16bf a, v16bf b, v8f c) {
  return __builtin_amdgcn_wmma_f32_16x16x32_bf16(
      /*neg_a=*/false, a, /*neg_b=*/false, b,
      /*c_mod=*/(short)0, c, /*reuse_a=*/false, /*reuse_b=*/false);
}

// ---------------------------------------------------------------------------
__global__ void cvt_f32_bf16_kernel(const float* __restrict__ src,
                                    bf16* __restrict__ dst, int nElem) {
  for (int i = blockIdx.x * blockDim.x + threadIdx.x; i < nElem;
       i += gridDim.x * blockDim.x)
    dst[i] = to_bf16(src[i]);
}

// ---------------------------------------------------------------------------
// C[M][N] = A[M][K] * W[N][K]^T + bias   (nn.Linear semantics)
// 128 threads = 4 waves; block tile 64x64; wave tile 16x64; K-step 64.
// ---------------------------------------------------------------------------
__global__ __launch_bounds__(128) void gemm_bias_kernel(
    const bf16* __restrict__ A, const bf16* __restrict__ W,
    const float* __restrict__ bias, void* __restrict__ outp,
    int M, int N, int K, int out_f32) {
  __shared__ __align__(16) bf16 Abuf[64 * 64];
  __shared__ __align__(16) bf16 Bbuf[64 * 64];

  const int tid  = threadIdx.x;
  const int w    = tid >> 5;
  const int lane = tid & 31;
  const int n    = lane & 15;
  const int hi   = lane >> 4;
  const int tM   = blockIdx.x * 64;
  const int tN   = blockIdx.y * 64;

  v8f acc[4];
#pragma unroll
  for (int c = 0; c < 4; ++c) acc[c] = zero_v8f();

  for (int k0 = 0; k0 < K; k0 += 64) {
    __syncthreads();
    // stage A[64][64] and W[64][64] bf16 tiles (16B chunks, async when avail)
#pragma unroll
    for (int ch = tid; ch < 512; ch += 128) {
      const int r = ch >> 3, c = (ch & 7) << 3;
      copy16_g2lds(A + (size_t)(tM + r) * K + k0 + c, Abuf + r * 64 + c);
      copy16_g2lds(W + (size_t)(tN + r) * K + k0 + c, Bbuf + r * 64 + c);
    }
    wait_async_lds();
    __syncthreads();

    const bf16* arow = Abuf + (w * 16 + n) * 64;
    const v16bf a0 = load_a_frag(arow, hi);
    const v16bf a1 = load_a_frag(arow + 32, hi);
#pragma unroll
    for (int c = 0; c < 4; ++c) {
      const bf16* brow = Bbuf + (c * 16 + n) * 64;
      const v16bf b0 = load_b_frag(brow, hi);
      const v16bf b1 = load_b_frag(brow + 32, hi);
      acc[c] = wmma_bf(a0, b0, acc[c]);
      acc[c] = wmma_bf(a1, b1, acc[c]);
    }
  }

  float* of = (float*)outp;
  bf16*  ob = (bf16*)outp;
#pragma unroll
  for (int c = 0; c < 4; ++c) {
#pragma unroll
    for (int v = 0; v < 8; ++v) {
      const int row = tM + w * 16 + v + 8 * hi;
      const int col = tN + c * 16 + n;
      const float val = acc[c][v] + bias[col];
      if (out_f32) of[(size_t)row * N + col] = val;
      else         ob[(size_t)row * N + col] = to_bf16(val);
    }
  }
}

// ---------------------------------------------------------------------------
// Flash attention per head. Block = 128 threads = 4 waves; each wave owns a
// 16-query-row tile (64 rows/block). Streams 32-key blocks through LDS.
// Mask: keep (i,j) iff (i%512) >= (j%512); softmax over full 2048 keys.
// ---------------------------------------------------------------------------
__global__ __launch_bounds__(128) void attn_fa_kernel(
    const bf16* __restrict__ Qb, const bf16* __restrict__ Kb,
    const bf16* __restrict__ Vb, bf16* __restrict__ Ob) {
  __shared__ __align__(16) bf16 Klds[32 * DKH];   // K rows j0..j0+31, 64 wide
  __shared__ __align__(16) bf16 VTlds[DKH * 32];  // V^T: [dk][32]
  __shared__ __align__(16) bf16 Plds[4][16 * 32]; // per-wave P tile

  const int h    = blockIdx.y;
  const int i0b  = blockIdx.x * 64;
  const int tid  = threadIdx.x;
  const int w    = tid >> 5;
  const int lane = tid & 31;
  const int n    = lane & 15;
  const int hi   = lane >> 4;
  const int i0   = i0b + w * 16;

  // Q tile (16x64) held in registers as two A-fragments.
  v16bf qa0, qa1;
  {
    const bf16* qrow = Qb + (size_t)(i0 + n) * DM + h * DKH;
    qa0 = load_a_frag(qrow, hi);
    qa1 = load_a_frag(qrow + 32, hi);
  }

  v8f accO[4];
#pragma unroll
  for (int c = 0; c < 4; ++c) accO[c] = zero_v8f();
  float rowM[8], rowL[8];
#pragma unroll
  for (int v = 0; v < 8; ++v) { rowM[v] = -1e30f; rowL[v] = 0.f; }

  const int imax_blk = (i0b & (WIN - 1)) + 63;
  const int imax_wav = (i0  & (WIN - 1)) + 15;

  for (int j0 = 0; j0 < L_SEQ; j0 += 32) {
    const int jm_lo = j0 & (WIN - 1);
    if (jm_lo > imax_blk) continue;  // uniform: whole block fully masked

    __syncthreads();  // previous iteration's LDS reads complete
    // stage K tile (async direct-to-LDS) and V^T tile (transpose scatter)
#pragma unroll
    for (int ch = tid; ch < 256; ch += 128) {
      const int r = ch >> 3, c = (ch & 7) << 3;
      copy16_g2lds(Kb + (size_t)(j0 + r) * DM + h * DKH + c,
                   Klds + r * DKH + c);
      const v8bf vv = *(const v8bf*)(Vb + (size_t)(j0 + r) * DM + h * DKH + c);
#pragma unroll
      for (int e = 0; e < 8; ++e) VTlds[(c + e) * 32 + r] = vv[e];
    }
    wait_async_lds();
    __syncthreads();

    const bool active = (jm_lo <= imax_wav);  // wave-uniform
    if (active) {
      // S = Q K^T  (two 16x16 tiles, K-depth 64 = 2 WMMA each)
      v8f s0 = zero_v8f(), s1 = zero_v8f();
      {
        const v16bf b00 = load_b_frag(Klds + n * DKH, hi);
        const v16bf b10 = load_b_frag(Klds + (16 + n) * DKH, hi);
        const v16bf b01 = load_b_frag(Klds + n * DKH + 32, hi);
        const v16bf b11 = load_b_frag(Klds + (16 + n) * DKH + 32, hi);
        s0 = wmma_bf(qa0, b00, s0);
        s1 = wmma_bf(qa0, b10, s1);
        s0 = wmma_bf(qa1, b01, s0);
        s1 = wmma_bf(qa1, b11, s1);
      }
      const int jmn0 = (j0 + n) & (WIN - 1);
      const int jmn1 = (j0 + 16 + n) & (WIN - 1);
#pragma unroll
      for (int v = 0; v < 8; ++v) {
        const int im = (i0 + v + 8 * hi) & (WIN - 1);
        const bool k0 = im >= jmn0;
        const bool k1 = im >= jmn1;
        float a0 = k0 ? s0[v] * 0.125f : -1e30f;  // 1/sqrt(64)
        float a1 = k1 ? s1[v] * 0.125f : -1e30f;
        float bm = fmaxf(a0, a1);
#pragma unroll
        for (int off = 1; off < 16; off <<= 1)
          bm = fmaxf(bm, __shfl_xor(bm, off, 32));  // row lives in 16 lanes
        const float Mn = fmaxf(rowM[v], bm);
        const float f  = __expf(rowM[v] - Mn);
        rowM[v] = Mn;
        const float e0 = k0 ? __expf(a0 - Mn) : 0.f;
        const float e1 = k1 ? __expf(a1 - Mn) : 0.f;
        float ps = e0 + e1;
#pragma unroll
        for (int off = 1; off < 16; off <<= 1)
          ps += __shfl_xor(ps, off, 32);
        rowL[v] = rowL[v] * f + ps;
#pragma unroll
        for (int c = 0; c < 4; ++c) accO[c][v] *= f;
        const int m = v + 8 * hi;
        Plds[w][m * 32 + n]      = to_bf16(e0);  // C-layout -> LDS row-major
        Plds[w][m * 32 + 16 + n] = to_bf16(e1);
      }
    }
    __syncthreads();  // P visible / ordered before re-read as A-operand
    if (active) {
      const v16bf pA = load_a_frag(&Plds[w][n * 32], hi);
#pragma unroll
      for (int c = 0; c < 4; ++c) {
        const v16bf bv = load_b_frag(VTlds + (c * 16 + n) * 32, hi);
        accO[c] = wmma_bf(pA, bv, accO[c]);
      }
    }
  }

  // normalize and store to merged [L][H*dk] bf16
#pragma unroll
  for (int c = 0; c < 4; ++c) {
#pragma unroll
    for (int v = 0; v < 8; ++v) {
      const int row = i0 + v + 8 * hi;
      const int col = h * DKH + c * 16 + n;
      Ob[(size_t)row * DM + col] = to_bf16(accO[c][v] / rowL[v]);
    }
  }
}

// ---------------------------------------------------------------------------
extern "C" void kernel_launch(void* const* d_in, const int* in_sizes, int n_in,
                              void* d_out, int out_size, void* d_ws,
                              size_t ws_size, hipStream_t stream) {
  const float* x  = (const float*)d_in[0];
  const float* Wq = (const float*)d_in[1];
  const float* bq = (const float*)d_in[2];
  const float* Wk = (const float*)d_in[3];
  const float* bk = (const float*)d_in[4];
  const float* Wv = (const float*)d_in[5];
  const float* bv = (const float*)d_in[6];
  const float* Wo = (const float*)d_in[7];
  const float* bo = (const float*)d_in[8];

  const size_t LD = (size_t)L_SEQ * DM;  // 2M elems
  const size_t DD = (size_t)DM * DM;     // 1M elems
  bf16* xb  = (bf16*)d_ws;     // workspace: ~28 MB of bf16 buffers
  bf16* Wqb = xb  + LD;
  bf16* Wkb = Wqb + DD;
  bf16* Wvb = Wkb + DD;
  bf16* Wob = Wvb + DD;
  bf16* Qb  = Wob + DD;
  bf16* Kb  = Qb  + LD;
  bf16* Vb  = Kb  + LD;
  bf16* Ab  = Vb  + LD;

  // 1) fp32 -> bf16 conversions
  cvt_f32_bf16_kernel<<<512, 256, 0, stream>>>(x,  xb,  (int)LD);
  cvt_f32_bf16_kernel<<<512, 256, 0, stream>>>(Wq, Wqb, (int)DD);
  cvt_f32_bf16_kernel<<<512, 256, 0, stream>>>(Wk, Wkb, (int)DD);
  cvt_f32_bf16_kernel<<<512, 256, 0, stream>>>(Wv, Wvb, (int)DD);
  cvt_f32_bf16_kernel<<<512, 256, 0, stream>>>(Wo, Wob, (int)DD);

  // 2) QKV projections (bf16 out, head-merged layout [L][H*64])
  dim3 gg(L_SEQ / 64, DM / 64);  // 32 x 16
  dim3 gb(128);
  gemm_bias_kernel<<<gg, gb, 0, stream>>>(xb, Wqb, bq, Qb, L_SEQ, DM, DM, 0);
  gemm_bias_kernel<<<gg, gb, 0, stream>>>(xb, Wkb, bk, Kb, L_SEQ, DM, DM, 0);
  gemm_bias_kernel<<<gg, gb, 0, stream>>>(xb, Wvb, bv, Vb, L_SEQ, DM, DM, 0);

  // 3) flash attention with tile mask
  dim3 ag(L_SEQ / 64, NH);  // 32 x 16
  attn_fa_kernel<<<ag, gb, 0, stream>>>(Qb, Kb, Vb, Ab);

  // 4) output projection, fp32 result straight to d_out
  gemm_bias_kernel<<<gg, gb, 0, stream>>>(Ab, Wob, bo, d_out, L_SEQ, DM, DM, 1);
}
